// Relative3DBias_64132451664502
// MI455X (gfx1250) — compile-verified, matrix operand order checked by reading, not run on hardware
//
#include <hip/hip_runtime.h>

// CDNA5 / gfx1250: wave32, WMMA f32_16x16x32_f16.
typedef __attribute__((ext_vector_type(16))) _Float16 v16h;
typedef __attribute__((ext_vector_type(8)))  float    v8f;

#define NB     2
#define NQ     1024
#define NK     1024
#define HIDDEN 64
#define HEADS  16

#define LOG2E      1.4426950408889634f
#define RBF_G2     81.151596f            // 56.25 * log2(e)
#define GELU_C2    2.3022087f            // 2 * 0.7978845608 * log2(e)
#define GELU_C3    0.10294816f           // GELU_C2 * 0.044715

// Branchless tanh-form GELU: x * (1 - 1/(exp(2s)+1)), s = 0.79788456(x + 0.044715 x^3)
__device__ __forceinline__ float gelu_fast(float x) {
  const float x2 = x * x;
  const float u  = x * (GELU_C2 + GELU_C3 * x2);      // 2s * log2(e)
  const float t  = __builtin_amdgcn_exp2f(u);         // v_exp_f32, no branches
  return x * (1.0f - __builtin_amdgcn_rcpf(t + 1.0f));
}

// One wave per (b, q, k-half). Each wave loops 32 k-tiles of 16 pairs.
// GEMM1: D1 = W1^T (A: 16 hidden x 32 feat) x feats^T (B: 32 x 16 pairs) + b1
// GEMM2: D2 = W2^T (A: 16 head x 64 hid)    x gelu(D1) (B)        -> lane = pair
// 16-bit operand K packing (ISA 7.12.2): element e of v16h, lane-half hb holds
// K(e) = (e&7) + 16*(e>>3) + 8*hb.  C/D: lane = N, row M = r + 8*hb.
__global__ __launch_bounds__(256)
void rel3d_bias_wmma(const float* __restrict__ qc,
                     const float* __restrict__ kc,
                     const float* __restrict__ W1,
                     const float* __restrict__ b1,
                     const float* __restrict__ W2,
                     float* __restrict__ out) {
  const int lane = threadIdx.x & 31;
  const int wid  = threadIdx.x >> 5;
  const int wg   = blockIdx.x * 8 + wid;          // 4096 waves total
  const int khalf = wg & 1;
  const int q     = (wg >> 1) & (NQ - 1);
  const int b     = wg >> 11;

  const int n  = lane & 15;                       // N index (pair / head-col)
  const int hb = lane >> 4;                       // lane half (compiler knows {0,1})

  // ---- persistent operands (registers) ----
  // A tiles for GEMM1: W1^T, 4 hidden-tiles of 16 x 32 (feature K, pad 20->32)
  v16h aW1[4];
  #pragma unroll
  for (int t = 0; t < 4; ++t) {
    #pragma unroll
    for (int e = 0; e < 16; ++e) {
      const int K = (e & 7) + 16 * (e >> 3) + 8 * hb;
      aW1[t][e] = (_Float16)((K < 20) ? W1[(size_t)K * HIDDEN + t * 16 + n] : 0.0f);
    }
  }
  // C bias tiles: c[t][r] = b1[t*16 + r + 8*hb]  (broadcast along pairs)
  v8f cb[4];
  #pragma unroll
  for (int t = 0; t < 4; ++t)
    #pragma unroll
    for (int r = 0; r < 8; ++r)
      cb[t][r] = b1[t * 16 + r + 8 * hb];
  // A tiles for GEMM2: W2^T, 16 heads x 64 hidden (two K-tiles of 32)
  v16h aW2[2];
  #pragma unroll
  for (int kb = 0; kb < 2; ++kb)
    #pragma unroll
    for (int e = 0; e < 16; ++e) {
      const int K = 32 * kb + (e & 7) + 16 * (e >> 3) + 8 * hb;
      aW2[kb][e] = (_Float16)W2[(size_t)K * HEADS + n];
    }

  // q center: uniform across the wave
  const float qx = qc[((size_t)b * NQ + q) * 3 + 0];
  const float qy = qc[((size_t)b * NQ + q) * 3 + 1];
  const float qz = qc[((size_t)b * NQ + q) * 3 + 2];

  const float kStep = 2.0f / 15.0f;   // RBF center spacing

  const int kt0 = khalf * 32;
  for (int kt = kt0; kt < kt0 + 32; ++kt) {
    const int kidx = kt * 16 + n;
    const float* kp = &kc[((size_t)b * NK + kidx) * 3];
    if (kt + 1 < kt0 + 32)
      __builtin_prefetch(kp + 48, 0, 1);          // next tile -> global_prefetch

    const float rx = qx - kp[0];
    const float ry = qy - kp[1];
    const float rz = qz - kp[2];
    const float dist = sqrtf(rx * rx + ry * ry + rz * rz);

    // feats^T in B-operand layout (lane = pair), fully branchless.
    // K(e) = (e&7) + 16*(e>>3) + 8*hb; features: 0-2 rel, 3 dist,
    // 4..19 rbf bins, >=20 zero. Known-bits on hb folds most selects.
    v16h fB;
    #pragma unroll
    for (int e = 0; e < 16; ++e) {
      const int K = (e & 7) + 16 * (e >> 3) + 8 * hb;
      const float c = (float)(K - 4) * kStep;
      const float d = dist - c;
      float f = __builtin_amdgcn_exp2f(-RBF_G2 * d * d);  // rbf(K-4)
      f = (K >= 20) ? 0.0f : f;
      f = (K == 3) ? dist : f;
      f = (K == 2) ? rz : f;
      f = (K == 1) ? ry : f;
      f = (K == 0) ? rx : f;
      fB[e] = (_Float16)f;
    }

    // GEMM1: h^T tiles (bias folded into C input)
    v8f d1[4];
    #pragma unroll
    for (int t = 0; t < 4; ++t)
      d1[t] = __builtin_amdgcn_wmma_f32_16x16x32_f16(
          false, aW1[t], false, fB, (short)0, cb[t], false, false);

    // GELU + f16 pack straight into GEMM2 B operand (pure in-lane repack:
    // C/D layout and 16-bit B layout share lane-half semantics).
    v16h b2[2];
    #pragma unroll
    for (int kb = 0; kb < 2; ++kb)
      #pragma unroll
      for (int v = 0; v < 8; ++v) {
        const int t = 2 * kb + (v >> 2);
        const int r = 2 * (v & 3);
        b2[kb][2 * v]     = (_Float16)gelu_fast(d1[t][r]);
        b2[kb][2 * v + 1] = (_Float16)gelu_fast(d1[t][r + 1]);
      }

    // GEMM2: chain K = 32 + 32
    v8f d2 = {};
    d2 = __builtin_amdgcn_wmma_f32_16x16x32_f16(
        false, aW2[0], false, b2[0], (short)0, d2, false, false);
    d2 = __builtin_amdgcn_wmma_f32_16x16x32_f16(
        false, aW2[1], false, b2[1], (short)0, d2, false, false);

    // out[b][head][q][k]: lane = pair -> 64B contiguous per half-wave row.
    // 8*tanh(x/8) = 8 - 16/(exp2(x*log2e/4)+1), branchless, saturates to +-8.
    #pragma unroll
    for (int r = 0; r < 8; ++r) {
      const int head = r + 8 * hb;
      const float t = __builtin_amdgcn_exp2f(d2[r] * (0.25f * LOG2E));
      const float val = 8.0f - 16.0f * __builtin_amdgcn_rcpf(t + 1.0f);
      out[(((size_t)b * HEADS + head) * NQ + q) * NK + kt * 16 + n] = val;
    }
  }
}

extern "C" void kernel_launch(void* const* d_in, const int* in_sizes, int n_in,
                              void* d_out, int out_size, void* d_ws, size_t ws_size,
                              hipStream_t stream) {
  const float* qc = (const float*)d_in[0];   // [2,1024,3]
  const float* kc = (const float*)d_in[1];   // [2,1024,3]
  const float* W1 = (const float*)d_in[2];   // [20,64]
  const float* b1 = (const float*)d_in[3];   // [64]
  const float* W2 = (const float*)d_in[4];   // [64,16]
  float* out = (float*)d_out;                // [2,16,1024,1024]

  // 4096 waves = 2(b) * 1024(q) * 2(k-half); 8 waves per 256-thread block
  rel3d_bias_wmma<<<dim3(512), dim3(256), 0, stream>>>(qc, kc, W1, b1, W2, out);
}
